// MultiHeadAttention_36429912605010
// MI455X (gfx1250) — matrix operand order repaired and checked by reading, not compiled, not run on hardware
//
#include <hip/hip_runtime.h>

// ---------------------------------------------------------------------------
// MI455X (gfx1250) wave32 WMMA implementation of 2D-RoPE multi-head attention.
// All GEMM stages use v_wmma_f32_16x16x32_f16 (f16 in, f32 accumulate).
// Wave tiles sized so each B-fragment load feeds two WMMAs (1.5 loads/WMMA).
// ---------------------------------------------------------------------------

typedef __attribute__((ext_vector_type(16))) _Float16 v16h;
typedef __attribute__((ext_vector_type(8)))  _Float16 v8h;
typedef __attribute__((ext_vector_type(8)))  float    v8f;
typedef __attribute__((ext_vector_type(4)))  float    v4f;

#define B_   2
#define T_   1024
#define DM   512
#define NH_  64
#define HD   256
#define KC   (NH_ * HD)   // 16384 concat dim

// ---- WMMA fragment helpers (wave32 layouts per CDNA5 ISA 7.12.2) ----------

// A-matrix 16x32 f16: lane m = l&15; K-base = (l>=16)?8:0; regs 0-7 = K kb..kb+7,
// regs 8-15 = K kb+16..kb+23.  Two contiguous 16B loads per lane.
__device__ inline v16h load_frag_a(const _Float16* __restrict__ A, int lda, int lane) {
  int m  = lane & 15;
  int kb = (lane >> 4) * 8;
  const _Float16* p = A + (size_t)m * lda + kb;
  v8h lo = *(const v8h*)(p);
  v8h hi = *(const v8h*)(p + 16);
  v16h r;
#pragma unroll
  for (int i = 0; i < 8; ++i) { r[i] = lo[i]; r[i + 8] = hi[i]; }
  return r;
}

// Same, but source is f32 (LDS-resident probabilities): b128 loads + cvt.
__device__ inline v16h load_frag_a_f32(const float* __restrict__ A, int lda, int lane) {
  int m  = lane & 15;
  int kb = (lane >> 4) * 8;
  const float* p = A + (size_t)m * lda + kb;
  v4f x0 = *(const v4f*)(p);
  v4f x1 = *(const v4f*)(p + 4);
  v4f x2 = *(const v4f*)(p + 16);
  v4f x3 = *(const v4f*)(p + 20);
  v16h r;
#pragma unroll
  for (int i = 0; i < 4; ++i) {
    r[i]      = (_Float16)x0[i];
    r[i + 4]  = (_Float16)x1[i];
    r[i + 8]  = (_Float16)x2[i];
    r[i + 12] = (_Float16)x3[i];
  }
  return r;
}

// B-matrix 32x16 f16 from N-major storage Bt[n][k]: lane n = l&15;
// K-base = (l>=16)?8:0; regs 0-7 = K kb..kb+7, regs 8-15 = K kb+16..kb+23.
__device__ inline v16h load_frag_bt(const _Float16* __restrict__ Bt, int ldb, int lane) {
  int n  = lane & 15;
  int kb = (lane >> 4) * 8;
  const _Float16* p = Bt + (size_t)n * ldb + kb;
  v8h lo = *(const v8h*)(p);
  v8h hi = *(const v8h*)(p + 16);
  v16h r;
#pragma unroll
  for (int i = 0; i < 8; ++i) { r[i] = lo[i]; r[i + 8] = hi[i]; }
  return r;
}

__device__ inline v8f wmma_f16(v16h a, v16h b, v8f c) {
  return __builtin_amdgcn_wmma_f32_16x16x32_f16(false, a, false, b, (short)0, c, false, false);
}

// ---- cast / transpose kernels ---------------------------------------------

__global__ void k_cvt_f16(const float* __restrict__ src, _Float16* __restrict__ dst, int n) {
  int i = blockIdx.x * blockDim.x + threadIdx.x;
  if (i < n) dst[i] = (_Float16)src[i];
}

// W [NH][DM][HD] f32 -> Wt [NH][HD][DM] f16  (N-major for B-fragments)
__global__ void k_transpose_w(const float* __restrict__ src, _Float16* __restrict__ dst) {
  int i = blockIdx.x * blockDim.x + threadIdx.x;
  if (i >= NH_ * DM * HD) return;
  int h   = i / (DM * HD);
  int rem = i % (DM * HD);
  int c   = rem / HD;   // contraction index
  int d   = rem % HD;   // output index
  dst[((size_t)h * HD + d) * DM + c] = (_Float16)src[i];
}

// Wp [KC][DM] f32 -> WpT [DM][KC] f16
__global__ void k_transpose_wp(const float* __restrict__ src, _Float16* __restrict__ dst) {
  int i = blockIdx.x * blockDim.x + threadIdx.x;
  if (i >= KC * DM) return;
  int kk = i / DM;
  int n  = i % DM;
  dst[(size_t)n * KC + kk] = (_Float16)src[i];
}

// ---- QKV projection + fused 2D RoPE ---------------------------------------
// grid: (T/64, NH, B*3)  block: 256 (8 waves, 2x4 -> 64 rows x 256 cols)
// Each wave: 32 rows x 64 cols -> each B-fragment feeds two WMMAs.
// mode 0 = Q (rope, [t][d]), 1 = K (rope, [t][d]), 2 = V (no rope, [d][t])

__global__ void k_proj(const _Float16* __restrict__ xb,
                       const _Float16* __restrict__ wqT,
                       const _Float16* __restrict__ wkT,
                       const _Float16* __restrict__ wvT,
                       const float* __restrict__ cosx, const float* __restrict__ sinx,
                       const float* __restrict__ cosy, const float* __restrict__ siny,
                       _Float16* __restrict__ q, _Float16* __restrict__ kOut,
                       _Float16* __restrict__ vT) {
  int lane = threadIdx.x & 31;
  int w    = threadIdx.x >> 5;
  int wr   = w >> 2, wc = w & 3;
  int m0   = blockIdx.x * 64 + wr * 32;
  int n0   = wc * 64;
  int h    = blockIdx.y;
  int mode = blockIdx.z % 3;
  int b    = blockIdx.z / 3;

  const _Float16* wT = (mode == 0 ? wqT : (mode == 1 ? wkT : wvT)) + (size_t)h * HD * DM;
  const _Float16* A  = xb + ((size_t)b * T_ + m0) * DM;

  v8f acc[2][4] = {};
  for (int kk = 0; kk < DM; kk += 32) {
    v16h a0 = load_frag_a(A + kk, DM, lane);
    v16h a1 = load_frag_a(A + (size_t)16 * DM + kk, DM, lane);
#pragma unroll
    for (int nt = 0; nt < 4; ++nt) {
      v16h bf = load_frag_bt(wT + (size_t)(n0 + nt * 16) * DM + kk, DM, lane);
      acc[0][nt] = wmma_f16(a0, bf, acc[0][nt]);
      acc[1][nt] = wmma_f16(a1, bf, acc[1][nt]);
    }
  }

  int half = lane >> 4;
  int nlo  = lane & 15;
  size_t bh = (size_t)b * NH_ + h;

  if (mode == 2) {
    _Float16* dst = vT + bh * (size_t)HD * T_;     // [d][t] transposed
#pragma unroll
    for (int mt = 0; mt < 2; ++mt)
#pragma unroll
      for (int nt = 0; nt < 4; ++nt)
#pragma unroll
        for (int j = 0; j < 8; ++j) {
          int t = m0 + mt * 16 + j + 8 * half;
          int d = n0 + nt * 16 + nlo;
          dst[(size_t)d * T_ + t] = (_Float16)acc[mt][nt][j];
        }
  } else {
    _Float16* dst = (mode == 0 ? q : kOut) + bh * (size_t)T_ * HD;
#pragma unroll
    for (int mt = 0; mt < 2; ++mt)
#pragma unroll
      for (int nt = 0; nt < 4; ++nt)
#pragma unroll
        for (int j = 0; j < 8; ++j) {
          int t = m0 + mt * 16 + j + 8 * half;
          int d = n0 + nt * 16 + nlo;
          float val = acc[mt][nt][j];
          float pv  = __shfl_xor(val, 1, 32);      // (re,im) partner lane
          const float* cp = (d < 128) ? cosx : cosy;
          const float* sp = (d < 128) ? sinx : siny;
          int f   = (d & 127) >> 1;
          float c = cp[t * 64 + f];
          float s = sp[t * 64 + f];
          float out = (d & 1) ? (pv * s + val * c) : (val * c - pv * s);
          dst[(size_t)t * HD + d] = (_Float16)out;
        }
  }
}

// ---- attention: scores + softmax + PV, per 16 query rows ------------------
// grid: (T/16, NH, B)  block: 256 (8 waves).  Dynamic LDS: S[16][1024] f32 (64KB).

__global__ void k_attn(const _Float16* __restrict__ q,
                       const _Float16* __restrict__ k,
                       const _Float16* __restrict__ vT,
                       _Float16* __restrict__ oc) {
  extern __shared__ char smem[];
  float* S = (float*)smem;                          // [16][T_]

  int lane = threadIdx.x & 31;
  int w    = threadIdx.x >> 5;
  int m0   = blockIdx.x * 16;
  int h    = blockIdx.y;
  int b    = blockIdx.z;
  size_t bh = (size_t)b * NH_ + h;

  const _Float16* Q = q  + bh * (size_t)T_ * HD + (size_t)m0 * HD;
  const _Float16* K = k  + bh * (size_t)T_ * HD;
  const _Float16* V = vT + bh * (size_t)HD * T_;    // [d][t]

  int half = lane >> 4;
  int nlo  = lane & 15;

  // Preload all 8 Q A-fragments for this row block (reused across 8 key tiles).
  v16h qa[8];
#pragma unroll
  for (int kd = 0; kd < 8; ++kd) qa[kd] = load_frag_a(Q + kd * 32, HD, lane);

  const float scale = 0.0441941738241592f;          // 1/sqrt(512) per reference

  // Pass 1: S = Q K^T * scale ; wave w covers keys [w*128, w*128+128)
  for (int kt = 0; kt < 8; ++kt) {
    int s0 = w * 128 + kt * 16;
    v8f acc = {};
#pragma unroll
    for (int kd = 0; kd < 8; ++kd) {
      v16h bf = load_frag_bt(K + (size_t)s0 * HD + kd * 32, HD, lane);
      acc = wmma_f16(qa[kd], bf, acc);
    }
#pragma unroll
    for (int j = 0; j < 8; ++j) {
      int m = j + 8 * half;
      S[m * T_ + s0 + nlo] = acc[j] * scale;
    }
  }
  __syncthreads();

  // Pass 2: softmax rows in place (16 threads per row, 64 elems each as v4f,
  // shfl_xor reductions stay inside a 16-lane group).
  {
    int r   = threadIdx.x >> 4;
    int seg = threadIdx.x & 15;
    float* row = S + r * T_ + seg * 64;
    v4f* rowv = (v4f*)row;
    float mx = -1e30f;
#pragma unroll 4
    for (int i = 0; i < 16; ++i) {
      v4f v = rowv[i];
      mx = fmaxf(fmaxf(fmaxf(mx, v[0]), fmaxf(v[1], v[2])), v[3]);
    }
#pragma unroll
    for (int msk = 1; msk < 16; msk <<= 1) mx = fmaxf(mx, __shfl_xor(mx, msk, 32));
    float ls = 0.f;
#pragma unroll 4
    for (int i = 0; i < 16; ++i) {
      v4f v = rowv[i];
      v[0] = __expf(v[0] - mx); v[1] = __expf(v[1] - mx);
      v[2] = __expf(v[2] - mx); v[3] = __expf(v[3] - mx);
      rowv[i] = v;
      ls += v[0] + v[1] + v[2] + v[3];
    }
#pragma unroll
    for (int msk = 1; msk < 16; msk <<= 1) ls += __shfl_xor(ls, msk, 32);
    float inv = 1.0f / ls;
#pragma unroll 4
    for (int i = 0; i < 16; ++i) {
      v4f v = rowv[i];
      v[0] *= inv; v[1] *= inv; v[2] *= inv; v[3] *= inv;
      rowv[i] = v;
    }
  }
  __syncthreads();

  // Pass 3: O = P x V ; wave w covers head-dim cols [w*32, w*32+32)
  int n0 = w * 32;
  v8f acc2[2] = {};
  for (int ss = 0; ss < T_; ss += 32) {
    v16h af = load_frag_a_f32(S + ss, T_, lane);
#pragma unroll
    for (int nt = 0; nt < 2; ++nt) {
      v16h bf = load_frag_bt(V + (size_t)(n0 + nt * 16) * T_ + ss, T_, lane);
      acc2[nt] = wmma_f16(af, bf, acc2[nt]);
    }
  }
#pragma unroll
  for (int nt = 0; nt < 2; ++nt)
#pragma unroll
    for (int j = 0; j < 8; ++j) {
      int t = m0 + j + 8 * half;
      int d = n0 + nt * 16 + nlo;
      oc[((size_t)b * T_ + t) * KC + h * HD + d] = (_Float16)acc2[nt][j];
    }
}

// ---- output projection: y = oc[2048,16384] @ Wp[16384,512] + bp -----------
// grid: ((B*T)/64, DM/256)  block: 256 (8 waves, 2x4 -> 64 rows x 256 cols)

__global__ void k_outproj(const _Float16* __restrict__ oc,
                          const _Float16* __restrict__ wpT,
                          const float* __restrict__ bp,
                          float* __restrict__ y) {
  int lane = threadIdx.x & 31;
  int w    = threadIdx.x >> 5;
  int wr   = w >> 2, wc = w & 3;
  int m0   = blockIdx.x * 64 + wr * 32;
  int n0   = blockIdx.y * 256 + wc * 64;

  const _Float16* A = oc + (size_t)m0 * KC;
  v8f acc[2][4] = {};
  for (int kk = 0; kk < KC; kk += 32) {
    // prefetch next K-slab of the streaming activation matrix
    __builtin_prefetch(A + kk + 64 + (size_t)(lane & 15) * KC, 0, 1);
    v16h a0 = load_frag_a(A + kk, KC, lane);
    v16h a1 = load_frag_a(A + (size_t)16 * KC + kk, KC, lane);
#pragma unroll
    for (int nt = 0; nt < 4; ++nt) {
      v16h bf = load_frag_bt(wpT + (size_t)(n0 + nt * 16) * KC + kk, KC, lane);
      acc[0][nt] = wmma_f16(a0, bf, acc[0][nt]);
      acc[1][nt] = wmma_f16(a1, bf, acc[1][nt]);
    }
  }
  int half = lane >> 4;
  int nlo  = lane & 15;
#pragma unroll
  for (int mt = 0; mt < 2; ++mt)
#pragma unroll
    for (int nt = 0; nt < 4; ++nt)
#pragma unroll
      for (int j = 0; j < 8; ++j) {
        int r = m0 + mt * 16 + j + 8 * half;
        int n = n0 + nt * 16 + nlo;
        y[(size_t)r * DM + n] = acc[mt][nt][j] + bp[n];
      }
}

// ---------------------------------------------------------------------------

extern "C" void kernel_launch(void* const* d_in, const int* in_sizes, int n_in,
                              void* d_out, int out_size, void* d_ws, size_t ws_size,
                              hipStream_t stream) {
  const float* x     = (const float*)d_in[0];
  const float* Wq    = (const float*)d_in[1];
  const float* Wk    = (const float*)d_in[2];
  const float* Wv    = (const float*)d_in[3];
  const float* Wp    = (const float*)d_in[4];
  const float* bp    = (const float*)d_in[5];
  const float* cos_x = (const float*)d_in[6];
  const float* sin_x = (const float*)d_in[7];
  const float* cos_y = (const float*)d_in[8];
  const float* sin_y = (const float*)d_in[9];
  float* y = (float*)d_out;

  char* ws = (char*)d_ws;
  size_t off = 0;
  auto carve = [&](size_t bytes) {
    size_t o = off;
    off = (off + bytes + 255) & ~(size_t)255;
    return o;
  };
  const size_t xb_b  = (size_t)B_ * T_ * DM * 2;
  const size_t wT_b  = (size_t)NH_ * HD * DM * 2;
  const size_t wpT_b = (size_t)DM * KC * 2;
  const size_t qk_b  = (size_t)B_ * NH_ * T_ * HD * 2;
  const size_t oc_b  = (size_t)B_ * T_ * KC * 2;

  _Float16* xb   = (_Float16*)(ws + carve(xb_b));
  _Float16* wqT  = (_Float16*)(ws + carve(wT_b));
  _Float16* wkT  = (_Float16*)(ws + carve(wT_b));
  _Float16* wvT  = (_Float16*)(ws + carve(wT_b));
  _Float16* wpT  = (_Float16*)(ws + carve(wpT_b));
  _Float16* qB   = (_Float16*)(ws + carve(qk_b));
  _Float16* kB   = (_Float16*)(ws + carve(qk_b));
  _Float16* vTB  = (_Float16*)(ws + carve(qk_b));
  _Float16* ocB  = (_Float16*)(ws + carve(oc_b));
  (void)ws_size; (void)in_sizes; (void)n_in; (void)out_size;

  // 1) casts / layout transforms
  {
    int n = B_ * T_ * DM;
    k_cvt_f16<<<(n + 255) / 256, 256, 0, stream>>>(x, xb, n);
    int nw = NH_ * DM * HD;
    k_transpose_w<<<(nw + 255) / 256, 256, 0, stream>>>(Wq, wqT);
    k_transpose_w<<<(nw + 255) / 256, 256, 0, stream>>>(Wk, wkT);
    k_transpose_w<<<(nw + 255) / 256, 256, 0, stream>>>(Wv, wvT);
    int np = KC * DM;
    k_transpose_wp<<<(np + 255) / 256, 256, 0, stream>>>(Wp, wpT);
  }

  // 2) QKV projection + RoPE (z: B*3 selects batch x {q,k,v})
  k_proj<<<dim3(T_ / 64, NH_, B_ * 3), 256, 0, stream>>>(
      xb, wqT, wkT, wvT, cos_x, sin_x, cos_y, sin_y, qB, kB, vTB);

  // 3) attention (64KB dynamic LDS for score rows)
  k_attn<<<dim3(T_ / 16, NH_, B_), 256, 16 * T_ * sizeof(float), stream>>>(
      qB, kB, vTB, ocB);

  // 4) output projection + bias
  k_outproj<<<dim3((B_ * T_) / 64, DM / 256), 256, 0, stream>>>(ocB, wpT, bp, y);
}
